// FFQLinear_33182917329331
// MI455X (gfx1250) — compile-verified, matrix operand order checked by reading, not run on hardware
//
#include <hip/hip_runtime.h>

typedef __attribute__((ext_vector_type(16))) _Float16 v16h;
typedef __attribute__((ext_vector_type(8)))  _Float16 v8h;
typedef __attribute__((ext_vector_type(4)))  _Float16 v4h;
typedef __attribute__((ext_vector_type(2)))  _Float16 v2h;
typedef __attribute__((ext_vector_type(8)))  float    v8f;
typedef __attribute__((ext_vector_type(4)))  float    v4f;
typedef __attribute__((ext_vector_type(4)))  int      v4i;

#define BM 128
#define BN 128
#define BK 32
#define LDK 40                 // padded K stride (halves): conflict-free b128 frag loads
#define A_SZ (BM * LDK)        // halves
#define B_SZ (BN * LDK)        // halves
#define STAGE (A_SZ + B_SZ)    // halves per double-buffer stage

__launch_bounds__(256, 2)
__global__ void ffq_gemm_wmma(const float* __restrict__ x,
                              const int*   __restrict__ q,
                              const float* __restrict__ scale,
                              const int*   __restrict__ zp,
                              const float* __restrict__ bias,
                              float*       __restrict__ out,
                              int M, int N, int K)
{
    __shared__ __align__(16) _Float16 smem[2 * STAGE];   // 40 KB

    const int tid = threadIdx.x;
    const int m0  = blockIdx.y * BM;
    const int n0  = blockIdx.x * BN;

    // ---- A loader indices: 128 rows x 32 K, float4 per load ----
    const int rowA0 = tid >> 3;    // 0..31, +32 per pass -> 128 rows
    const int cgA   = tid & 7;     // float4 group within 32-wide K strip

    // ---- B loader indices: k-pair units so converted halves pack to b32 ----
    const int kp0 = tid >> 5;      // 0..7, +8 per pass -> 16 k-pairs (32 k-rows)
    const int ngB = tid & 31;      // int4 group within 128-wide N strip

    // per-thread zero-points for its 4 fixed B columns
    const v4i zp4 = *(const v4i*)(zp + n0 + ngB * 4);

    // ---- compute indices ----
    const int w     = tid >> 5;
    const int lane  = tid & 31;
    const int wm    = w >> 1;      // 0..3 -> 32 rows each
    const int wn    = w & 1;       // 0..1 -> 64 cols each
    const int m16   = lane & 15;
    const int khalf = lane >> 4;

    v8f c[2][4];
#pragma unroll
    for (int mt = 0; mt < 2; ++mt)
#pragma unroll
        for (int nt = 0; nt < 4; ++nt)
            c[mt][nt] = v8f{};

    v4f aReg[4];
    v4i bLo[2], bHi[2];

    const int NK = K / BK;

    auto loadG = [&](int kt) {
        const float* ap = x + (size_t)(m0 + rowA0) * K + (size_t)kt * BK + cgA * 4;
#pragma unroll
        for (int p = 0; p < 4; ++p)
            aReg[p] = *(const v4f*)(ap + (size_t)(p * 32) * K);
#pragma unroll
        for (int p = 0; p < 2; ++p) {
            const int kpair = kp0 + 8 * p;                    // 0..15
            const int* bp = q + (size_t)((size_t)kt * BK + 2 * kpair) * N + n0 + ngB * 4;
            bLo[p] = *(const v4i*)(bp);
            bHi[p] = *(const v4i*)(bp + N);
        }
    };

    auto storeL = [&](int buf) {
        _Float16* As = smem + buf * STAGE;
        _Float16* Bs = As + A_SZ;
#pragma unroll
        for (int p = 0; p < 4; ++p) {
            v4h h;
#pragma unroll
            for (int j = 0; j < 4; ++j) h[j] = (_Float16)aReg[p][j];
            *(v4h*)(As + (rowA0 + 32 * p) * LDK + cgA * 4) = h;   // ds_store_b64
        }
#pragma unroll
        for (int p = 0; p < 2; ++p) {
            const int kpair = kp0 + 8 * p;
#pragma unroll
            for (int j = 0; j < 4; ++j) {    // transpose to N-major, fold -zp, pack k-pair
                v2h hv;
                hv[0] = (_Float16)(short)(bLo[p][j] - zp4[j]);
                hv[1] = (_Float16)(short)(bHi[p][j] - zp4[j]);
                *(v2h*)(Bs + (ngB * 4 + j) * LDK + 2 * kpair) = hv;  // ds_store_b32
            }
        }
    };

    auto compute = [&](int buf) {
        const _Float16* As = smem + buf * STAGE;
        const _Float16* Bs = As + A_SZ;
        v16h a[2], b[4];
#pragma unroll
        for (int mt = 0; mt < 2; ++mt) {
            const _Float16* p = As + (wm * 32 + mt * 16 + m16) * LDK + 8 * khalf;
            v8h lo = *(const v8h*)(p);        // K = 8*khalf + 0..7
            v8h hi = *(const v8h*)(p + 16);   // K = 16 + 8*khalf + 0..7
            a[mt] = __builtin_shufflevector(lo, hi, 0,1,2,3,4,5,6,7,8,9,10,11,12,13,14,15);
        }
#pragma unroll
        for (int nt = 0; nt < 4; ++nt) {
            const _Float16* p = Bs + (wn * 64 + nt * 16 + m16) * LDK + 8 * khalf;
            v8h lo = *(const v8h*)(p);
            v8h hi = *(const v8h*)(p + 16);
            b[nt] = __builtin_shufflevector(lo, hi, 0,1,2,3,4,5,6,7,8,9,10,11,12,13,14,15);
        }
#pragma unroll
        for (int mt = 0; mt < 2; ++mt)
#pragma unroll
            for (int nt = 0; nt < 4; ++nt)
                c[mt][nt] = __builtin_amdgcn_wmma_f32_16x16x32_f16(
                    false, a[mt], false, b[nt], (short)0, c[mt][nt], false, false);
    };

    loadG(0);
    storeL(0);
    __syncthreads();

    // steady-state loop: branch-free body, one barrier per K-tile
    for (int kt = 0; kt < NK - 1; ++kt) {
        loadG(kt + 1);
        {   // clamped speculative prefetch into GL2 (global_prefetch_b8)
            const int kp = (kt + 2 < NK) ? (kt + 2) : (NK - 1);
            __builtin_prefetch(x + (size_t)(m0 + rowA0) * K + (size_t)kp * BK + cgA * 4, 0, 0);
            __builtin_prefetch(q + (size_t)((size_t)kp * BK + 2 * kp0) * N + n0 + ngB * 4, 0, 0);
        }
        compute(kt & 1);
        storeL((kt + 1) & 1);
        __syncthreads();
    }
    compute((NK - 1) & 1);

    // ---- epilogue: out = acc * scale[n] + bias[n] ----
#pragma unroll
    for (int nt = 0; nt < 4; ++nt) {
        const int   n  = n0 + wn * 64 + nt * 16 + m16;
        const float s  = scale[n];
        const float bb = bias[n];
#pragma unroll
        for (int mt = 0; mt < 2; ++mt) {
            const int mg = m0 + wm * 32 + mt * 16 + khalf * 8;
            float* op = out + (size_t)mg * N + n;
#pragma unroll
            for (int r = 0; r < 8; ++r)
                op[(size_t)r * N] = c[mt][nt][r] * s + bb;
        }
    }
}

extern "C" void kernel_launch(void* const* d_in, const int* in_sizes, int n_in,
                              void* d_out, int out_size, void* d_ws, size_t ws_size,
                              hipStream_t stream) {
    const float* x     = (const float*)d_in[0];
    const int*   q     = (const int*)  d_in[1];
    const float* scale = (const float*)d_in[2];
    const int*   zp    = (const int*)  d_in[3];
    const float* bias  = (const float*)d_in[4];
    float*       out   = (float*)d_out;

    const int D = in_sizes[2];           // 4096 (scale length = N = K)
    const int M = in_sizes[0] / D;       // B*S = 4096

    dim3 grid(D / BN, M / BM);           // 32 x 32 blocks
    ffq_gemm_wmma<<<grid, 256, 0, stream>>>(x, q, scale, zp, bias, out, M, D, D);
}
